// FusedDecoder_66005057404945
// MI455X (gfx1250) — compile-verified, matrix operand order checked by reading, not run on hardware
//
#include <hip/hip_runtime.h>
#include <hip/hip_bf16.h>
#include <math.h>

#ifndef M_PI
#define M_PI 3.14159265358979323846
#endif

// ---------------------------------------------------------------------------
// CDNA5 WMMA types
// ---------------------------------------------------------------------------
typedef __attribute__((ext_vector_type(16))) _Float16 v16h;
typedef __attribute__((ext_vector_type(8)))  float    v8f;

// A fragment from a 32-wide row-major f16 LDS tile (pointer pre-offset to the
// wave's 16-row sub-tile). Layout (ISA 7.12.2, 16-bit A 16x32):
// lane<16: row=lane, K in {0..7,16..23}; lane>=16: row=lane-16, K in {8..15,24..31}.
__device__ inline v16h load_frag_a32(const _Float16* s) {
  int lane = threadIdx.x & 31;
  int r = lane & 15;
  int base = (lane < 16) ? 0 : 8;
  v16h f;
#pragma unroll
  for (int j = 0; j < 16; ++j) {
    int kk = base + j + ((j >= 8) ? 8 : 0);
    f[j] = s[r * 32 + kk];
  }
  return f;
}
// B fragment from a 32-wide row-major [k][n] f16 LDS tile (pointer pre-offset
// to the wave's 16-column sub-tile); mirrored layout.
__device__ inline v16h load_frag_b32(const _Float16* s) {
  int lane = threadIdx.x & 31;
  int c = lane & 15;
  int base = (lane < 16) ? 0 : 8;
  v16h f;
#pragma unroll
  for (int j = 0; j < 16; ++j) {
    int kk = base + j + ((j >= 8) ? 8 : 0);
    f[j] = s[kk * 32 + c];
  }
  return f;
}

// gfx1250 async global->LDS DMA (ASYNCcnt-tracked), inline asm so it works on
// both toolchains regardless of builtin arity.
__device__ inline void async_ld_b64(void* lds, const void* g) {
  asm volatile("global_load_async_to_lds_b64 %0, %1, off"
               :: "v"((unsigned)(size_t)lds), "v"(g) : "memory");
}
__device__ inline void wait_async0() {
  asm volatile("s_wait_asynccnt 0x0" ::: "memory");
}

// ---------------------------------------------------------------------------
// Generic batched complex GEMM:  C[b] = alpha * A[b] * B[b]  (+= if ACC)
//   AREAL/BREAL: operand is real float (else interleaved float2).
//   Strides in *elements of that array's type* (float2 counts as 1 for complex).
//   Batch b = b1*nB2 + b2;  per-operand batch strides sX1 (b1), sX2 (b2).
//   4 waves / block: 32x32 C macro-tile, each wave owns a 16x16 sub-tile;
//   one cooperative LDS stage feeds all four waves' v_wmma_f32_16x16x32_f16.
// ---------------------------------------------------------------------------
template <int AREAL, int BREAL, int ACC>
__global__ __launch_bounds__(128) void cgemm_wmma(
    const float* __restrict__ Abase, const float* __restrict__ Bbase,
    float* __restrict__ Cbase, int M, int N, int K, int rsA, int csA,
    long long sA1, long long sA2, int rsB, int csB, long long sB1,
    long long sB2, int rsC, long long sC1, long long sC2, int nB2,
    float alpha) {
  const int b = blockIdx.z;
  const int b1 = b / nB2, b2 = b % nB2;
  const int m0 = blockIdx.y * 32, n0 = blockIdx.x * 32;
  const int tid = threadIdx.x;
  const int wv = tid >> 5;
  const int moff = (wv >> 1) << 4;   // wave's row sub-tile
  const int noff = (wv & 1) << 4;    // wave's col sub-tile

  __shared__ _Float16 Asr[32 * 32];
  __shared__ _Float16 Asi[32 * 32];
  __shared__ _Float16 Bsr[32 * 32];
  __shared__ _Float16 Bsi[32 * 32];

  v8f rr = {}, ii = {}, ri = {}, ir = {};

  const long long offA = b1 * sA1 + b2 * sA2;
  const long long offB = b1 * sB1 + b2 * sB2;
  const float2* A2p = (const float2*)Abase;
  const float2* B2p = (const float2*)Bbase;

  for (int k0 = 0; k0 < K; k0 += 32) {
    // stage A (32 rows x 32 k) with zero padding
    for (int idx = tid; idx < 32 * 32; idx += 128) {
      int r = idx >> 5, kk = idx & 31;
      int gm = m0 + r, gk = k0 + kk;
      float ar = 0.f, ai = 0.f;
      if (gm < M && gk < K) {
        long long a = offA + (long long)gm * rsA + (long long)gk * csA;
        if (AREAL) {
          ar = Abase[a];
        } else {
          float2 v = A2p[a];
          ar = v.x; ai = v.y;
        }
      }
      Asr[idx] = (_Float16)ar;
      Asi[idx] = (_Float16)ai;
    }
    // stage B (32 k x 32 cols)
    for (int idx = tid; idx < 32 * 32; idx += 128) {
      int kk = idx >> 5, c = idx & 31;
      int gk = k0 + kk, gn = n0 + c;
      float br = 0.f, bi = 0.f;
      if (gk < K && gn < N) {
        long long a = offB + (long long)gk * rsB + (long long)gn * csB;
        if (BREAL) {
          br = Bbase[a];
        } else {
          float2 v = B2p[a];
          br = v.x; bi = v.y;
        }
      }
      Bsr[idx] = (_Float16)br;
      Bsi[idx] = (_Float16)bi;
    }
    // prefetch next A chunk (-> global_prefetch_b8)
    if (k0 + 32 < K) {
      long long a = offA + (long long)m0 * rsA + (long long)(k0 + 32) * csA;
      __builtin_prefetch(AREAL ? (const void*)(Abase + a) : (const void*)(A2p + a), 0, 1);
    }
    __syncthreads();

    v16h aR = load_frag_a32(Asr + moff * 32);
    v16h bR = load_frag_b32(Bsr + noff);
    rr = __builtin_amdgcn_wmma_f32_16x16x32_f16(false, aR, false, bR, (short)0,
                                                rr, false, false);
    if (!BREAL) {
      v16h bI = load_frag_b32(Bsi + noff);
      ri = __builtin_amdgcn_wmma_f32_16x16x32_f16(false, aR, false, bI,
                                                  (short)0, ri, false, false);
      if (!AREAL) {
        v16h aI = load_frag_a32(Asi + moff * 32);
        ii = __builtin_amdgcn_wmma_f32_16x16x32_f16(false, aI, false, bI,
                                                    (short)0, ii, false, false);
        ir = __builtin_amdgcn_wmma_f32_16x16x32_f16(false, aI, false, bR,
                                                    (short)0, ir, false, false);
      }
    } else if (!AREAL) {
      v16h aI = load_frag_a32(Asi + moff * 32);
      ir = __builtin_amdgcn_wmma_f32_16x16x32_f16(false, aI, false, bR,
                                                  (short)0, ir, false, false);
    }
    __syncthreads();
  }

  // C layout (ISA 7.12.2): lane<16: N=lane, M=vgpr; lane>=16: N=lane-16, M=vgpr+8
  const int lane = tid & 31;
  float2* Cp = (float2*)Cbase + b1 * sC1 + b2 * sC2;
#pragma unroll
  for (int v = 0; v < 8; ++v) {
    int m = m0 + moff + v + ((lane & 16) ? 8 : 0);
    int n = n0 + noff + (lane & 15);
    if (m < M && n < N) {
      float cr = alpha * (rr[v] - ii[v]);
      float ci = alpha * (ri[v] + ir[v]);
      float2* p = Cp + (long long)m * rsC + n;
      if (ACC) {
        p->x += cr;
        p->y += ci;
      } else {
        *p = make_float2(cr, ci);
      }
    }
  }
}

// ---------------------------------------------------------------------------
// Constant builders
// ---------------------------------------------------------------------------
__global__ void k_zero(float* __restrict__ p, long long n) {
  long long i = (long long)blockIdx.x * 256 + threadIdx.x;
  if (i < n) p[i] = 0.f;
}

// DFT / phase matrix: msOnRows: out[r][c] = cis(sign*2pi*(r-(L-1))*c/n)  (F)
//                     else:      out[r][c] = cis(sign*2pi*(c-(L-1))*r/n)  (G)
__global__ void k_dft(float2* __restrict__ out, int rows, int cols, int L,
                      int n, int sign, int msOnRows) {
  long long i = (long long)blockIdx.x * 256 + threadIdx.x;
  if (i >= (long long)rows * cols) return;
  int r = (int)(i / cols), c = (int)(i % cols);
  int msv = (msOnRows ? r : c) - (L - 1);
  int ui = msOnRows ? c : r;
  double ph = (double)sign * 2.0 * M_PI * (double)msv * (double)ui / (double)n;
  out[i] = make_float2((float)cos(ph), (float)sin(ph));
}

__global__ void k_softw(float* __restrict__ wq, int b) {
  int j = blockIdx.x * blockDim.x + threadIdx.x;
  if (j >= 2 * b) return;
  double s = 0.0;
  for (int t = 0; t < b; ++t)
    s += sin((2.0 * j + 1) * (2.0 * t + 1) * M_PI / (4.0 * b)) / (2.0 * t + 1);
  wq[j] = (float)(2.0 / b * sin(M_PI * (2.0 * j + 1) / (4.0 * b)) * s);
}

__device__ inline double dpowi(double x, int e) {
  double r = 1.0, p = x;
  while (e > 0) {
    if (e & 1) r *= p;
    p *= p;
    e >>= 1;
  }
  return r;
}

// Wigner small-d via the explicit factorial sum (double precision).
__device__ double wigd(int l, int mp, int m, double beta, const double* fact) {
  double cb = cos(0.5 * beta), sb = sin(0.5 * beta);
  int smin = max(0, m - mp), smax = min(l + m, l - mp);
  if (smin > smax) return 0.0;
  double pref =
      sqrt(fact[l + mp] * fact[l - mp] * fact[l + m] * fact[l - m]);
  double sum = 0.0;
  for (int s = smin; s <= smax; ++s) {
    double den = fact[l + m - s] * fact[s] * fact[mp - m + s] * fact[l - mp - s];
    double term = dpowi(cb, 2 * l + m - mp - 2 * s) * dpowi(sb, mp - m + 2 * s) / den;
    sum += ((mp - m + s) & 1) ? -term : term;
  }
  return pref * sum;
}

// Fills A (wq * d at beta_in), S ((2l+1) * d at beta_out), Dg (phase * d at grid beta).
// grid = (L, Kin + Kinv + 1). Grid: 6 alphas x 1 beta x 6 gammas (g = i*6+j).
__global__ __launch_bounds__(256) void k_wigner(
    int L, int Kin, int Kinv, double betaGrid, const float* __restrict__ wq,
    float* __restrict__ A, float* __restrict__ S, float2* __restrict__ Dg) {
  const int l = blockIdx.x;
  const int task = blockIdx.y;
  const int Mdim = 2 * L - 1;
  const int MM = Mdim * Mdim;
  const int tid = threadIdx.x;
  __shared__ double fact[64];
  if (tid == 0) {
    fact[0] = 1.0;
    for (int i = 1; i < 64; ++i) fact[i] = fact[i - 1] * i;
  }
  __syncthreads();
  const int w = 2 * l + 1;
  double beta;
  int mode, j = 0;
  if (task < Kin) {
    mode = 0; j = task;
    beta = M_PI * (2.0 * j + 1) / (2.0 * Kin);
  } else if (task < Kin + Kinv) {
    mode = 1; j = task - Kin;
    beta = M_PI * (2.0 * j + 1) / (2.0 * Kinv);
  } else {
    mode = 2;
    beta = betaGrid;
  }
  if (mode == 0 && A == nullptr) return;
  if (mode == 2 && Dg == nullptr) return;
  for (int p = tid; p < w * w; p += 256) {
    int mp = p / w - l;
    int mN = p % w - l;
    double d = wigd(l, mp, mN, beta, fact);
    int mi = (L - 1) + mp, ni = (L - 1) + mN;
    if (mode == 0) {
      A[(long long)(l * Kin + j) * MM + mi * Mdim + ni] = (float)(wq[j] * d);
    } else if (mode == 1) {
      S[(long long)(l * Kinv + j) * MM + mi * Mdim + ni] =
          (float)((2 * l + 1) * d);
    } else {
      for (int g = 0; g < 36; ++g) {
        int gi = g / 6, gj = g % 6;
        double alpha = 2.0 * M_PI * gi / 6.0;
        double gamma = (-2.0 * M_PI + (4.0 * M_PI / 6.0) * gj) - alpha;
        double ph = -(alpha * mp + gamma * mN);
        Dg[(long long)(l * 36 + g) * MM + mi * Mdim + ni] =
            make_float2((float)(d * cos(ph)), (float)(d * sin(ph)));
      }
    }
  }
}

// ---------------------------------------------------------------------------
// Pointwise / contraction helpers
// ---------------------------------------------------------------------------
// X[c][l][mn] = sum_k A[l][k][mn] * xf[c][k][mn]
__global__ void k_xA(const float* __restrict__ A, const float2* __restrict__ xf,
                     float2* __restrict__ X, int Ci, int L, int Kb, int MM) {
  long long i = (long long)blockIdx.x * 256 + threadIdx.x;
  long long total = (long long)Ci * L * MM;
  if (i >= total) return;
  int mn = (int)(i % MM);
  long long t = i / MM;
  int l = (int)(t % L);
  int c = (int)(t / L);
  float sr = 0.f, si = 0.f;
  for (int k = 0; k < Kb; ++k) {
    float a = A[((long long)l * Kb + k) * MM + mn];
    float2 v = xf[((long long)c * Kb + k) * MM + mn];
    sr += a * v.x;
    si += a * v.y;
  }
  X[((long long)c * L + l) * MM + mn] = make_float2(sr, si);
}

// yf[o][k][mn] = sum_l S[l][k][mn] * Y[l][o][mn]
__global__ void k_yS(const float* __restrict__ S, const float2* __restrict__ Y,
                     float2* __restrict__ yf, int Co, int Kinv, int L, int MM) {
  long long i = (long long)blockIdx.x * 256 + threadIdx.x;
  long long total = (long long)Co * Kinv * MM;
  if (i >= total) return;
  int mn = (int)(i % MM);
  long long t = i / MM;
  int k = (int)(t % Kinv);
  int o = (int)(t / Kinv);
  float sr = 0.f, si = 0.f;
  for (int l = 0; l < L; ++l) {
    float s = S[((long long)l * Kinv + k) * MM + mn];
    float2 y = Y[((long long)l * Co + o) * MM + mn];
    sr += s * y.x;
    si += s * y.y;
  }
  yf[((long long)o * Kinv + k) * MM + mn] = make_float2(sr, si);
}

// BN statistics from frequency domain: mean = DC coeff, E[y^2] via Parseval.
__global__ __launch_bounds__(256) void k_bn_stats(
    const float2* __restrict__ yf, int K, int MM, int c0,
    const float* __restrict__ g, const float* __restrict__ b,
    float* __restrict__ scale, float* __restrict__ shift) {
  int o = blockIdx.x, tid = threadIdx.x;
  const float2* base = yf + (long long)o * K * MM;
  float s1 = 0.f, s2 = 0.f;
  for (long long i = tid; i < (long long)K * MM; i += 256) {
    float2 v = base[i];
    s2 += v.x * v.x + v.y * v.y;
    if ((int)(i % MM) == c0) s1 += v.x;
  }
  __shared__ float r1[256], r2[256];
  r1[tid] = s1;
  r2[tid] = s2;
  __syncthreads();
  for (int s = 128; s > 0; s >>= 1) {
    if (tid < s) {
      r1[tid] += r1[tid + s];
      r2[tid] += r2[tid + s];
    }
    __syncthreads();
  }
  if (tid == 0) {
    float mean = r1[0] / K;
    float var = r2[0] / K - mean * mean;
    float sc = g[o] * rsqrtf(var + 1e-5f);
    scale[o] = sc;
    shift[o] = b[o] - mean * sc;
  }
}

// h = prelu(bn(real part of tmp)))
__global__ void k_pointwise(const float2* __restrict__ tmp,
                            const float* __restrict__ scale,
                            const float* __restrict__ shift,
                            const float* __restrict__ pa,
                            float* __restrict__ h, long long perC,
                            long long total) {
  long long i = (long long)blockIdx.x * 256 + threadIdx.x;
  if (i >= total) return;
  int c = (int)(i / perC);
  float z = tmp[i].x * scale[c] + shift[c];
  h[i] = (z >= 0.f) ? z : pa[0] * z;
}

// x2 = concat(e_img_lidar, upsample5(bn1(y1)))
__global__ void k_gather(const float* __restrict__ e,
                         const float2* __restrict__ y1,
                         const float* __restrict__ sc1,
                         const float* __restrict__ sh1,
                         float* __restrict__ x2) {
  long long i = (long long)blockIdx.x * 256 + threadIdx.x;
  if (i >= 90ll * 125000) return;
  int c = (int)(i / 125000);
  int rem = (int)(i % 125000);
  int k = rem / 2500, r2 = rem % 2500, u = r2 / 50, v = r2 % 50;
  float val;
  if (c < 60) {
    val = e[(long long)c * 125000 + rem];
  } else {
    int cc = c - 60;
    float yr = y1[(((long long)cc * 10 + k / 5) * 10 + u / 5) * 10 + v / 5].x;
    val = yr * sc1[cc] + sh1[cc];
  }
  x2[i] = val;
}

// Final: per (o,k) plane, reconstruct y[u][v] from yf3, apply bn3+prelu3,
// reduce over v (gamma), times 2*pi/200.  Never materializes the 200^3 tensor.
// The yf3 plane is staged into LDS with the CDNA5 async global->LDS DMA.
__global__ __launch_bounds__(256) void k_final(
    const float2* __restrict__ yf3, const float2* __restrict__ G3,
    const float* __restrict__ scale, const float* __restrict__ shift,
    const float* __restrict__ pa, float* __restrict__ out) {
  const int o = blockIdx.y;   // 9
  const int k = blockIdx.x;   // 200
  const int tid = threadIdx.x;
  constexpr int M = 49, NV = 200;
  __shared__ float2 yfl[M * M];       // 19208 B
  __shared__ _Float16 tr[M * NV];     // 19600 B
  __shared__ _Float16 ti[M * NV];     // 19600 B
  __shared__ float2 grow[M];          // 392 B
  __shared__ float red[256];          // 1024 B

  const float2* yfp = yf3 + ((long long)o * 200 + k) * (M * M);
  // async DMA: 2401 x 8-byte chunks, ASYNCcnt-tracked
  for (int i = tid; i < M * M; i += 256)
    async_ld_b64(&yfl[i], &yfp[i]);
  wait_async0();
  __syncthreads();

  // t[m][v] = sum_n yfl[m][n] * G3[v][n]
  for (int i = tid; i < M * NV; i += 256) {
    int m = i / NV, v = i - m * NV;
    float sr = 0.f, si = 0.f;
    const float2* yr = &yfl[m * M];
    const float2* gv = &G3[(long long)v * M];
    for (int n = 0; n < M; ++n) {
      float2 a = yr[n], g = gv[n];
      sr += a.x * g.x - a.y * g.y;
      si += a.x * g.y + a.y * g.x;
    }
    tr[i] = (_Float16)sr;
    ti[i] = (_Float16)si;
  }
  __syncthreads();

  const float sc = scale[o], sh = shift[o], a2 = pa[0];
  for (int u = 0; u < 200; ++u) {
    if (tid < M) grow[tid] = G3[(long long)u * M + tid];
    __syncthreads();
    float acc = 0.f;
    if (tid < NV) {
      int v = tid;
      float yr = 0.f;
      for (int m = 0; m < M; ++m) {
        float2 g = grow[m];
        yr += g.x * (float)tr[m * NV + v] - g.y * (float)ti[m * NV + v];
      }
      float z = yr * sc + sh;
      acc = (z >= 0.f) ? z : a2 * z;
    }
    red[tid] = acc;
    __syncthreads();
    for (int s = 128; s > 0; s >>= 1) {
      if (tid < s) red[tid] += red[tid + s];
      __syncthreads();
    }
    if (tid == 0)
      out[((long long)o * 200 + k) * 200 + u] =
          red[0] * (float)(2.0 * M_PI / 200.0);
    __syncthreads();
  }
}

// ---------------------------------------------------------------------------
// Host orchestration
// ---------------------------------------------------------------------------
static void launch_cgemm(int AR, int BR, int ACC, dim3 grid, hipStream_t st,
                         const void* A, const void* B, void* C, int M, int N,
                         int K, int rsA, int csA, long long sA1, long long sA2,
                         int rsB, int csB, long long sB1, long long sB2,
                         int rsC, long long sC1, long long sC2, int nB2,
                         float alpha) {
  const float* Af = (const float*)A;
  const float* Bf = (const float*)B;
  float* Cf = (float*)C;
  if (AR == 0 && BR == 1 && ACC == 0)
    cgemm_wmma<0, 1, 0><<<grid, 128, 0, st>>>(Af, Bf, Cf, M, N, K, rsA, csA,
                                              sA1, sA2, rsB, csB, sB1, sB2,
                                              rsC, sC1, sC2, nB2, alpha);
  else if (AR == 0 && BR == 0 && ACC == 0)
    cgemm_wmma<0, 0, 0><<<grid, 128, 0, st>>>(Af, Bf, Cf, M, N, K, rsA, csA,
                                              sA1, sA2, rsB, csB, sB1, sB2,
                                              rsC, sC1, sC2, nB2, alpha);
  else if (AR == 1 && BR == 0 && ACC == 0)
    cgemm_wmma<1, 0, 0><<<grid, 128, 0, st>>>(Af, Bf, Cf, M, N, K, rsA, csA,
                                              sA1, sA2, rsB, csB, sB1, sB2,
                                              rsC, sC1, sC2, nB2, alpha);
  else
    cgemm_wmma<0, 0, 1><<<grid, 128, 0, st>>>(Af, Bf, Cf, M, N, K, rsA, csA,
                                              sA1, sA2, rsB, csB, sB1, sB2,
                                              rsC, sC1, sC2, nB2, alpha);
}

static inline dim3 g1d(long long n) { return dim3((unsigned)((n + 255) / 256)); }

extern "C" void kernel_launch(void* const* d_in, const int* in_sizes, int n_in,
                              void* d_out, int out_size, void* d_ws,
                              size_t ws_size, hipStream_t stream) {
  const float* x   = (const float*)d_in[0];   // (80,10,10,10)
  const float* e   = (const float*)d_in[1];   // (60,50,50,50)
  const float* k1  = (const float*)d_in[2];   // (80,30,36)
  const float* b1g = (const float*)d_in[3];
  const float* b1b = (const float*)d_in[4];
  const float* k2  = (const float*)d_in[5];   // (90,90,36)
  const float* b2g = (const float*)d_in[6];
  const float* b2b = (const float*)d_in[7];
  const float* pa2 = (const float*)d_in[8];
  const float* k3  = (const float*)d_in[9];   // (90,9,36)
  const float* b3g = (const float*)d_in[10];
  const float* b3b = (const float*)d_in[11];
  const float* pa3 = (const float*)d_in[12];
  float* out = (float*)d_out;

  // bump allocator over d_ws (needs ~450 MB)
  char* base = (char*)d_ws;
  size_t off = 0;
  auto allocf = [&](size_t n) -> float* {
    float* p = (float*)(base + off);
    off = (off + n * 4 + 255) & ~(size_t)255;
    return p;
  };
  auto allocc = [&](size_t n) -> float2* {
    float2* p = (float2*)(base + off);
    off = (off + n * 8 + 255) & ~(size_t)255;
    return p;
  };

  // constants
  float2* F1 = allocc(9 * 10);
  float2* G1 = allocc(10 * 9);
  float2* F2 = allocc(49 * 50);
  float2* G2 = allocc(50 * 49);
  float2* G3 = allocc(200 * 49);
  float* wq1 = allocf(10);
  float* wq2 = allocf(50);
  float* A1 = allocf(5 * 10 * 81);
  float* S1 = allocf(5 * 10 * 81);
  float2* Dg1 = allocc(5 * 36 * 81);
  float* A2 = allocf((size_t)25 * 50 * 2401);
  float* S2 = allocf((size_t)25 * 50 * 2401);
  float2* Dg2 = allocc((size_t)25 * 36 * 2401);
  float* S3 = allocf((size_t)25 * 200 * 2401);
  // conv1 intermediates
  float2* t1a = allocc(80 * 10 * 90);
  float2* xf1 = allocc(80 * 10 * 81);
  float2* X1 = allocc(80 * 405);
  float2* Z1 = allocc(30 * 405);
  float2* Y1 = allocc(5 * 30 * 81);
  float2* yf1 = allocc(30 * 10 * 81);
  float2* ti1 = allocc(30 * 10 * 90);
  float2* y1c = allocc(30 * 10 * 100);
  float* sc1 = allocf(30); float* sh1 = allocf(30);
  float* sc2 = allocf(90); float* sh2 = allocf(90);
  float* sc3 = allocf(9);  float* sh3 = allocf(9);
  // big buffers (aliased across stages)
  float* x2h = allocf((size_t)90 * 125000);           // x2, then h
  float2* B1 = allocc((size_t)90 * 50 * 2500);        // xf / yf / inv-out / yf3
  float2* B2 = allocc((size_t)90 * 50 * 49 * 50);     // t1 / t2
  float2* Xb = allocc((size_t)90 * 60025);
  float2* Zb = allocc((size_t)90 * 60025);
  float2* Yb = allocc((size_t)25 * 90 * 2401);

  // ---- constants ----
  k_zero<<<g1d(5 * 10 * 81), 256, 0, stream>>>(A1, 5 * 10 * 81);
  k_zero<<<g1d(5 * 10 * 81), 256, 0, stream>>>(S1, 5 * 10 * 81);
  k_zero<<<g1d(2ll * 5 * 36 * 81), 256, 0, stream>>>((float*)Dg1, 2ll * 5 * 36 * 81);
  k_zero<<<g1d(25ll * 50 * 2401), 256, 0, stream>>>(A2, 25ll * 50 * 2401);
  k_zero<<<g1d(25ll * 50 * 2401), 256, 0, stream>>>(S2, 25ll * 50 * 2401);
  k_zero<<<g1d(2ll * 25 * 36 * 2401), 256, 0, stream>>>((float*)Dg2, 2ll * 25 * 36 * 2401);
  k_zero<<<g1d(25ll * 200 * 2401), 256, 0, stream>>>(S3, 25ll * 200 * 2401);
  k_softw<<<1, 64, 0, stream>>>(wq1, 5);
  k_softw<<<1, 64, 0, stream>>>(wq2, 25);
  k_dft<<<g1d(90), 256, 0, stream>>>(F1, 9, 10, 5, 10, -1, 1);
  k_dft<<<g1d(90), 256, 0, stream>>>(G1, 10, 9, 5, 10, +1, 0);
  k_dft<<<g1d(2450), 256, 0, stream>>>(F2, 49, 50, 25, 50, -1, 1);
  k_dft<<<g1d(2450), 256, 0, stream>>>(G2, 50, 49, 25, 50, +1, 0);
  k_dft<<<g1d(9800), 256, 0, stream>>>(G3, 200, 49, 25, 200, +1, 0);
  k_wigner<<<dim3(5, 21), 256, 0, stream>>>(5, 10, 10, M_PI / 32.0, wq1, A1, S1, Dg1);
  k_wigner<<<dim3(25, 101), 256, 0, stream>>>(25, 50, 50, M_PI / 64.0, wq2, A2, S2, Dg2);
  k_wigner<<<dim3(25, 201), 256, 0, stream>>>(25, 0, 200, 0.0, nullptr, nullptr, S3, nullptr);

  const float scale1 = 1.0f / sqrtf(36.0f * 80.0f);
  const float scale23 = 1.0f / sqrtf(36.0f * 90.0f);

  // ---- conv1 (b=5, M=9, n=10, Ci=80, Co=30) ----
  launch_cgemm(0, 1, 0, dim3(1, 1, 800), stream, F1, x, t1a, 9, 10, 10,
               10, 1, 0, 0, 10, 1, 1000, 100, 10, 900, 90, 10, 1.0f);
  launch_cgemm(0, 0, 0, dim3(1, 1, 800), stream, t1a, F1, xf1, 9, 9, 10,
               10, 1, 900, 90, 1, 10, 0, 0, 9, 810, 81, 10, 1.0f / 100.0f);
  k_xA<<<g1d(80ll * 5 * 81), 256, 0, stream>>>(A1, xf1, X1, 80, 5, 10, 81);
  k_zero<<<g1d(2ll * 5 * 30 * 81), 256, 0, stream>>>((float*)Y1, 2ll * 5 * 30 * 81);
  for (int g = 0; g < 36; ++g) {
    launch_cgemm(1, 0, 0, dim3(13, 1, 1), stream, k1 + g, X1, Z1, 30, 405, 80,
                 36, 1080, 0, 0, 405, 1, 0, 0, 405, 0, 0, 1, scale1);
    launch_cgemm(0, 0, 1, dim3(1, 1, 150), stream, Z1, Dg1 + (size_t)g * 81,
                 Y1, 9, 9, 9, 9, 1, 81, 405, 9, 1, 2916, 0, 9, 2430, 81, 30,
                 1.0f);
  }
  k_yS<<<g1d(30ll * 10 * 81), 256, 0, stream>>>(S1, Y1, yf1, 30, 10, 5, 81);
  k_bn_stats<<<30, 256, 0, stream>>>(yf1, 10, 81, 40, b1g, b1b, sc1, sh1);
  launch_cgemm(0, 0, 0, dim3(1, 1, 300), stream, G1, yf1, ti1, 10, 9, 9,
               9, 1, 0, 0, 9, 1, 810, 81, 9, 900, 90, 10, 1.0f);
  launch_cgemm(0, 0, 0, dim3(1, 1, 300), stream, ti1, G1, y1c, 10, 10, 9,
               9, 1, 900, 90, 1, 9, 0, 0, 10, 1000, 100, 10, 1.0f);

  // ---- conv2 (b=25, M=49, n=50, Ci=90, Co=90) ----
  k_gather<<<g1d(90ll * 125000), 256, 0, stream>>>(e, y1c, sc1, sh1, x2h);
  launch_cgemm(0, 1, 0, dim3(2, 2, 4500), stream, F2, x2h, B2, 49, 50, 50,
               50, 1, 0, 0, 50, 1, 125000, 2500, 50, 122500, 2450, 50, 1.0f);
  launch_cgemm(0, 0, 0, dim3(2, 2, 4500), stream, B2, F2, B1, 49, 49, 50,
               50, 1, 122500, 2450, 1, 50, 0, 0, 49, 120050, 2401, 50,
               1.0f / 2500.0f);
  k_xA<<<g1d(90ll * 25 * 2401), 256, 0, stream>>>(A2, B1, Xb, 90, 25, 50, 2401);
  k_zero<<<g1d(2ll * 25 * 90 * 2401), 256, 0, stream>>>((float*)Yb, 2ll * 25 * 90 * 2401);
  for (int g = 0; g < 36; ++g) {
    launch_cgemm(1, 0, 0, dim3(1876, 3, 1), stream, k2 + g, Xb, Zb, 90, 60025,
                 90, 36, 3240, 0, 0, 60025, 1, 0, 0, 60025, 0, 0, 1, scale23);
    launch_cgemm(0, 0, 1, dim3(2, 2, 2250), stream, Zb, Dg2 + (size_t)g * 2401,
                 Yb, 49, 49, 49, 49, 1, 2401, 60025, 49, 1, 86436, 0, 49,
                 216090, 2401, 90, 1.0f);
  }
  k_yS<<<g1d(90ll * 50 * 2401), 256, 0, stream>>>(S2, Yb, B1, 90, 50, 25, 2401);
  k_bn_stats<<<90, 256, 0, stream>>>(B1, 50, 2401, 1200, b2g, b2b, sc2, sh2);
  launch_cgemm(0, 0, 0, dim3(2, 2, 4500), stream, G2, B1, B2, 50, 49, 49,
               49, 1, 0, 0, 49, 1, 120050, 2401, 49, 122500, 2450, 50, 1.0f);
  launch_cgemm(0, 0, 0, dim3(2, 2, 4500), stream, B2, G2, B1, 50, 50, 49,
               49, 1, 122500, 2450, 1, 49, 0, 0, 50, 125000, 2500, 50, 1.0f);
  k_pointwise<<<g1d(90ll * 125000), 256, 0, stream>>>(B1, sc2, sh2, pa2, x2h,
                                                      125000, 90ll * 125000);

  // ---- conv3 (b_in=25, b_inv=100, Ci=90, Co=9) ----
  launch_cgemm(0, 1, 0, dim3(2, 2, 4500), stream, F2, x2h, B2, 49, 50, 50,
               50, 1, 0, 0, 50, 1, 125000, 2500, 50, 122500, 2450, 50, 1.0f);
  launch_cgemm(0, 0, 0, dim3(2, 2, 4500), stream, B2, F2, B1, 49, 49, 50,
               50, 1, 122500, 2450, 1, 50, 0, 0, 49, 120050, 2401, 50,
               1.0f / 2500.0f);
  k_xA<<<g1d(90ll * 25 * 2401), 256, 0, stream>>>(A2, B1, Xb, 90, 25, 50, 2401);
  k_zero<<<g1d(2ll * 25 * 9 * 2401), 256, 0, stream>>>((float*)Yb, 2ll * 25 * 9 * 2401);
  for (int g = 0; g < 36; ++g) {
    launch_cgemm(1, 0, 0, dim3(1876, 1, 1), stream, k3 + g, Xb, Zb, 9, 60025,
                 90, 36, 324, 0, 0, 60025, 1, 0, 0, 60025, 0, 0, 1, scale23);
    launch_cgemm(0, 0, 1, dim3(2, 2, 225), stream, Zb, Dg2 + (size_t)g * 2401,
                 Yb, 49, 49, 49, 49, 1, 2401, 60025, 49, 1, 86436, 0, 49,
                 21609, 2401, 9, 1.0f);
  }
  k_yS<<<g1d(9ll * 200 * 2401), 256, 0, stream>>>(S3, Yb, B1, 9, 200, 25, 2401);
  k_bn_stats<<<9, 256, 0, stream>>>(B1, 200, 2401, 1200, b3g, b3b, sc3, sh3);
  k_final<<<dim3(200, 9), 256, 0, stream>>>(B1, G3, sc3, sh3, pa3, out);
}
